// GAT_65532611002921
// MI455X (gfx1250) — matrix operand order repaired and checked by reading, not compiled, not run on hardware
//
#include <hip/hip_runtime.h>
#include <math.h>

typedef __attribute__((ext_vector_type(16))) _Float16 v16h;
typedef __attribute__((ext_vector_type(8)))  _Float16 v8h;
typedef __attribute__((ext_vector_type(8)))  float    v8f;

#define LSTR 144   // LDS row stride in halves: 288B, multiple of 32B -> aligned v16h rows

// ---------------------------------------------------------------------------
// Generic C = A(Mx128, fp32) * W(128 x ncols, fp32) + bias  via f16 WMMA,
// fp32 accumulate.  A rows come from A0 (rows < M0) or A1 (rows >= M0).
// Block: 256 threads (8 waves) computes 64 rows x 128 cols.
// ---------------------------------------------------------------------------
__global__ __launch_bounds__(256)
void k_gemm_wmma(const float* __restrict__ A0, long M0,
                 const float* __restrict__ A1,
                 long M,
                 const float* __restrict__ W,
                 const float* __restrict__ bias,
                 float* __restrict__ C,
                 int ncols)
{
    __shared__ _Float16 sW[128][LSTR];   // sW[n][k]  (transposed weights)
    __shared__ _Float16 sA[64][LSTR];    // sA[r][k]

    const int tid  = threadIdx.x;
    const int wave = tid >> 5;
    const int lane = tid & 31;

    // Stage weights (row-major [k][n] fp32 -> transposed f16 in LDS)
    const int wtot = 128 * ncols;
    for (int idx = tid; idx < wtot; idx += 256) {
        int k = idx / ncols;
        int n = idx - k * ncols;
        sW[n][k] = (_Float16)W[idx];
    }

    // Stage A tile (64 rows x 128 cols)
    const long row0 = (long)blockIdx.x * 64;
    for (int idx = tid; idx < 64 * 128; idx += 256) {
        int r = idx >> 7;
        int k = idx & 127;
        long gr = row0 + r;
        float v = 0.0f;
        if (gr < M)
            v = (gr < M0) ? A0[gr * 128 + k] : A1[(gr - M0) * 128 + k];
        sA[r][k] = (_Float16)v;
    }
    __syncthreads();

    const int ctile = wave * 16;
    if (ctile >= ncols) return;

    v8f acc[4];
#pragma unroll
    for (int rt = 0; rt < 4; ++rt)
#pragma unroll
        for (int i = 0; i < 8; ++i) acc[rt][i] = 0.0f;

    const int nloc = lane & 15;
    const int hi16 = lane >> 4;

#pragma unroll
    for (int kk = 0; kk < 128; kk += 32) {
        // B fragment: col = ctile+nloc; lanes 0-15: K=kk..kk+15, lanes 16-31: +16
        v16h bf = *(const v16h*)&sW[ctile + nloc][kk + (hi16 << 4)];
#pragma unroll
        for (int rt = 0; rt < 4; ++rt) {
            // A fragment: row = rt*16 + nloc; K runs {kb..kb+15}
            const int kb = kk + (hi16 << 4);
            v16h af = *(const v16h*)&sA[rt * 16 + nloc][kb];
            acc[rt] = __builtin_amdgcn_wmma_f32_16x16x32_f16(
                false, af, false, bf, (short)0, acc[rt], false, false);
        }
    }

    const int col = ctile + nloc;
    const float b = bias ? bias[col] : 0.0f;
#pragma unroll
    for (int rt = 0; rt < 4; ++rt) {
#pragma unroll
        for (int r = 0; r < 8; ++r) {
            long gr = row0 + rt * 16 + hi16 * 8 + r;   // C/D layout: VGPR r -> M=r (+8 upper half)
            if (gr < M) C[gr * ncols + col] = acc[rt][r] + b;
        }
    }
}

// ---------------------------------------------------------------------------
__global__ void k_fill(float* __restrict__ p, long n, float v)
{
    long i = (long)blockIdx.x * blockDim.x + threadIdx.x;
    if (i < n) p[i] = v;
}

// deg + sum of incoming edge_attr (for self-loop mean).  32 threads per edge.
__global__ void k_degloop(const float* __restrict__ ea, const int* __restrict__ dst,
                          long E, float* __restrict__ deg, float* __restrict__ loopa)
{
    long t = (long)blockIdx.x * blockDim.x + threadIdx.x;
    if (t >= E * 32) return;
    long e = t >> 5;
    int  q = (int)(t & 31);
    int  d = dst[e];
    const float* pe = ea + e * 128 + q * 4;
    float* pl = loopa + (long)d * 128 + q * 4;
    atomicAdd(&pl[0], pe[0]);
    atomicAdd(&pl[1], pe[1]);
    atomicAdd(&pl[2], pe[2]);
    atomicAdd(&pl[3], pe[3]);
    if (q == 0) atomicAdd(&deg[d], 1.0f);
}

__global__ void k_loopdiv(float* __restrict__ loopa, const float* __restrict__ deg, long N)
{
    long t = (long)blockIdx.x * blockDim.x + threadIdx.x;
    if (t >= N * 128) return;
    long n = t >> 7;
    loopa[t] /= fmaxf(deg[n], 1.0f);
}

__device__ inline void atomicMaxFloat(float* addr, float val)
{
    int cur = __float_as_int(*addr);
    while (__int_as_float(cur) < val) {
        int prev = atomicCAS((int*)addr, cur, __float_as_int(val));
        if (prev == cur) break;
        cur = prev;
    }
}

// Per (edge, head): leaky-relu message dotted with att -> logit; segment max.
__global__ void k_logit(const float* __restrict__ xl, const float* __restrict__ xr,
                        const float* __restrict__ ee,
                        const int* __restrict__ src, const int* __restrict__ dst,
                        long E, long M, const float* __restrict__ att,
                        float* __restrict__ logit, float* __restrict__ mx)
{
    long t = (long)blockIdx.x * blockDim.x + threadIdx.x;
    if (t >= M * 8) return;
    long e = t >> 3;
    int  h = (int)(t & 7);
    int  s = (e < E) ? src[e] : (int)(e - E);
    int  d = (e < E) ? dst[e] : (int)(e - E);
    const float* pl = xl + (long)s * 128 + h * 16;
    const float* pr = xr + (long)d * 128 + h * 16;
    const float* pe = ee + e * 128 + h * 16;
    float acc = 0.0f;
#pragma unroll
    for (int c = 0; c < 16; ++c) {
        float m = pl[c] + pr[c] + pe[c];
        m = (m > 0.0f) ? m : 0.2f * m;
        acc += m * att[h * 16 + c];
    }
    logit[e * 8 + h] = acc;
    atomicMaxFloat(&mx[(long)d * 8 + h], acc);
}

__global__ void k_exp(float* __restrict__ logit, const float* __restrict__ mx,
                      float* __restrict__ ssum,
                      const int* __restrict__ dst, long E, long M)
{
    long t = (long)blockIdx.x * blockDim.x + threadIdx.x;
    if (t >= M * 8) return;
    long e = t >> 3;
    int  h = (int)(t & 7);
    int  d = (e < E) ? dst[e] : (int)(e - E);
    float p = expf(logit[e * 8 + h] - mx[(long)d * 8 + h]);
    logit[e * 8 + h] = p;
    atomicAdd(&ssum[(long)d * 8 + h], p);
}

__global__ void k_agg(const float* __restrict__ xl, const float* __restrict__ p,
                      const float* __restrict__ ssum,
                      const int* __restrict__ src, const int* __restrict__ dst,
                      long E, long M, float* __restrict__ outb)
{
    long t = (long)blockIdx.x * blockDim.x + threadIdx.x;
    if (t >= M * 8) return;
    long e = t >> 3;
    int  h = (int)(t & 7);
    int  s = (e < E) ? src[e] : (int)(e - E);
    int  d = (e < E) ? dst[e] : (int)(e - E);
    float a = p[e * 8 + h] / ssum[(long)d * 8 + h];
    const float* pl = xl + (long)s * 128 + h * 16;
    float* po = outb + (long)d * 128 + h * 16;
#pragma unroll
    for (int c = 0; c < 16; ++c)
        atomicAdd(&po[c], pl[c] * a);
}

__global__ __launch_bounds__(256)
void k_bnstat(const float* __restrict__ z, long n, float* __restrict__ gsum)
{
    __shared__ float sh_s[128], sh_q[128];
    int tid = threadIdx.x;
    if (tid < 128) { sh_s[tid] = 0.0f; sh_q[tid] = 0.0f; }
    __syncthreads();
    int col  = tid & 127;
    int half = tid >> 7;
    float s = 0.0f, q = 0.0f;
    for (long r = (long)blockIdx.x * 2 + half; r < n; r += (long)gridDim.x * 2) {
        float v = z[r * 128 + col];
        s += v; q += v * v;
    }
    atomicAdd(&sh_s[col], s);
    atomicAdd(&sh_q[col], q);
    __syncthreads();
    if (tid < 128) {
        atomicAdd(&gsum[tid], sh_s[tid]);
        atomicAdd(&gsum[128 + tid], sh_q[tid]);
    }
}

__global__ void k_bnapply(const float* __restrict__ z, const float* __restrict__ gsum,
                          const float* __restrict__ w, const float* __restrict__ b,
                          float* __restrict__ x, long n, int residual)
{
    long t = (long)blockIdx.x * blockDim.x + threadIdx.x;
    if (t >= n * 128) return;
    int col = (int)(t & 127);
    float inv_n = 1.0f / (float)n;
    float mu  = gsum[col] * inv_n;
    float var = gsum[128 + col] * inv_n - mu * mu;
    float v = (z[t] - mu) * rsqrtf(var + 1e-5f) * w[col] + b[col];
    float g = 0.5f * v * (1.0f + erff(v * 0.70710678118654752f));
    x[t] = residual ? (g + x[t]) : g;
}

__global__ void k_gelu(float* __restrict__ p, long n)
{
    long t = (long)blockIdx.x * blockDim.x + threadIdx.x;
    if (t >= n) return;
    float v = p[t];
    p[t] = 0.5f * v * (1.0f + erff(v * 0.70710678118654752f));
}

// ---------------------------------------------------------------------------
extern "C" void kernel_launch(void* const* d_in, const int* in_sizes, int n_in,
                              void* d_out, int out_size, void* d_ws, size_t ws_size,
                              hipStream_t stream)
{
    (void)n_in; (void)out_size; (void)ws_size;

    const float* x_in = (const float*)d_in[0];
    const int*   ei   = (const int*)d_in[1];
    const float* ea   = (const float*)d_in[2];
    const float* Wl   = (const float*)d_in[3];
    const float* bl   = (const float*)d_in[4];
    const float* Wr   = (const float*)d_in[5];
    const float* br   = (const float*)d_in[6];
    const float* We   = (const float*)d_in[7];
    const float* att  = (const float*)d_in[8];
    // d_in[9] bias_o: zeros and cancels in BN mean subtraction -> omitted
    const float* bnw  = (const float*)d_in[10];
    const float* bnb  = (const float*)d_in[11];
    const float* W1   = (const float*)d_in[12];
    const float* b1   = (const float*)d_in[13];
    const float* W2   = (const float*)d_in[14];
    const float* b2   = (const float*)d_in[15];

    const long N = in_sizes[0] / 128;
    const long E = in_sizes[1] / 2;
    const long M = E + N;
    const int* src = ei;
    const int* dst = ei + E;

    float* ws    = (float*)d_ws;
    float* xbuf  = ws;  ws += N * 128;
    float* xl    = ws;  ws += N * 128;
    float* xr    = ws;  ws += N * 128;
    float* loopa = ws;  ws += N * 128;
    float* deg   = ws;  ws += N;
    float* mx    = ws;  ws += N * 8;
    float* ssum  = ws;  ws += N * 8;
    float* gsum  = ws;  ws += 256;
    float* logit = ws;  ws += M * 8;
    float* ee    = ws;  ws += M * 128;
    float* outb  = ee;   // alias: ee dead after k_logit

    auto nb = [](long t) { return (unsigned)((t + 255) / 256); };
    const unsigned gN = (unsigned)((N + 63) / 64);
    const unsigned gM = (unsigned)((M + 63) / 64);

    hipMemcpyAsync(xbuf, x_in, (size_t)N * 128 * sizeof(float),
                   hipMemcpyDeviceToDevice, stream);

    // self-loop attrs = mean of incoming edge attrs
    k_fill<<<nb(N), 256, 0, stream>>>(deg, N, 0.0f);
    k_fill<<<nb(N * 128), 256, 0, stream>>>(loopa, N * 128, 0.0f);
    k_degloop<<<nb(E * 32), 256, 0, stream>>>(ea, dst, E, deg, loopa);
    k_loopdiv<<<nb(N * 128), 256, 0, stream>>>(loopa, deg, N);

    for (int l = 0; l < 3; ++l) {
        const float* Wl_i = Wl + (long)l * 16384;
        const float* Wr_i = Wr + (long)l * 16384;
        const float* We_i = We + (long)l * 16384;

        k_gemm_wmma<<<gN, 256, 0, stream>>>(xbuf, N, xbuf, N, Wl_i, bl + l * 128, xl, 128);
        k_gemm_wmma<<<gN, 256, 0, stream>>>(xbuf, N, xbuf, N, Wr_i, br + l * 128, xr, 128);
        k_gemm_wmma<<<gM, 256, 0, stream>>>(ea, E, loopa, M, We_i, nullptr, ee, 128);

        k_fill<<<nb(N * 8), 256, 0, stream>>>(mx, N * 8, -3.0e38f);
        k_fill<<<nb(N * 8), 256, 0, stream>>>(ssum, N * 8, 0.0f);
        k_logit<<<nb(M * 8), 256, 0, stream>>>(xl, xr, ee, src, dst, E, M,
                                               att + (long)l * 128, logit, mx);
        k_fill<<<nb(N * 128), 256, 0, stream>>>(outb, N * 128, 0.0f);
        k_exp<<<nb(M * 8), 256, 0, stream>>>(logit, mx, ssum, dst, E, M);
        k_agg<<<nb(M * 8), 256, 0, stream>>>(xl, logit, ssum, src, dst, E, M, outb);

        k_fill<<<1, 256, 0, stream>>>(gsum, 256, 0.0f);
        k_bnstat<<<256, 256, 0, stream>>>(outb, N, gsum);
        k_bnapply<<<nb(N * 128), 256, 0, stream>>>(outb, gsum, bnw + l * 128,
                                                   bnb + l * 128, xbuf, N, l > 0);
    }

    // final MLP: Linear(128,128) -> GELU -> Linear(128,64)
    k_gemm_wmma<<<gN, 256, 0, stream>>>(xbuf, N, xbuf, N, W1, b1, xl, 128);
    k_gelu<<<nb(N * 128), 256, 0, stream>>>(xl, N * 128);
    k_gemm_wmma<<<gN, 256, 0, stream>>>(xl, N, xl, N, W2, b2, (float*)d_out, 64);
}
